// AttrClassifier_55602646614714
// MI455X (gfx1250) — matrix operand order repaired and checked by reading, not compiled, not run on hardware
//
#include <hip/hip_runtime.h>
#include <hip/hip_bf16.h>

typedef __attribute__((ext_vector_type(16))) __bf16 v16bf;
typedef __attribute__((ext_vector_type(8)))  __bf16 v8bf;
typedef __attribute__((ext_vector_type(8)))  float  v8f;

// Problem sizes (fixed by the reference)
constexpr int BSZ = 512;    // batch
constexpr int CSZ = 600;    // num classes
constexpr int DSZ = 25088;  // feature dim (K)

constexpr long FEAT_N = (long)BSZ * DSZ;  // 12,845,056
constexpr long W_N    = (long)CSZ * DSZ;  // 15,052,800

constexpr int KSPLIT = 8;
constexpr int KCHUNK = DSZ / KSPLIT;  // 3136, multiple of 64
constexpr int NPAD   = 640;          // padded N (5 blocks * 128)
constexpr int BM = 128, BN = 128;

// ---------------------------------------------------------------------------
// Stage 0: one-shot fp32 -> bf16 conversion (8 elems/thread, exact grid).
// ---------------------------------------------------------------------------
__global__ __launch_bounds__(256)
void cvt_f32_bf16(const float* __restrict__ src, __bf16* __restrict__ dst) {
  const long i = ((long)blockIdx.x * 256 + threadIdx.x) * 8;
  const float4* p = (const float4*)(src + i);
  float4 a = p[0], b = p[1];
  v8bf r;
  r[0] = (__bf16)a.x; r[1] = (__bf16)a.y; r[2] = (__bf16)a.z; r[3] = (__bf16)a.w;
  r[4] = (__bf16)b.x; r[5] = (__bf16)b.y; r[6] = (__bf16)b.z; r[7] = (__bf16)b.w;
  *(v8bf*)(dst + i) = r;
}

// ---------------------------------------------------------------------------
// Fragment loaders (bf16 source, wave32 WMMA layouts, CDNA5 ISA §7.12.2)
// `base` is wave-uniform (feat/W + k0); `off` is a fixed 32-bit per-lane
// element offset; the k-step lands in the 24-bit immediate.
// A (16x32 MxK): per-lane 8 contiguous K at +0 and 8 more at +16 elems.
// B (32x16 KxN): per-lane 8 contiguous K at +0 and 8 more at +16 of one W row (col of B).
// ---------------------------------------------------------------------------
__device__ __forceinline__ v16bf load_frag_a(const __bf16* __restrict__ base,
                                             int off) {
  v8bf lo = *(const v8bf*)(base + off);        // K +0..7
  v8bf hi = *(const v8bf*)(base + off + 16);   // K +16..23
  return __builtin_shufflevector(lo, hi, 0,1,2,3,4,5,6,7,8,9,10,11,12,13,14,15);
}
__device__ __forceinline__ v16bf load_frag_b(const __bf16* __restrict__ base,
                                             int off) {
  v8bf lo = *(const v8bf*)(base + off);        // K +0..7
  v8bf hi = *(const v8bf*)(base + off + 16);   // K +16..23
  return __builtin_shufflevector(lo, hi, 0,1,2,3,4,5,6,7,8,9,10,11,12,13,14,15);
}

// ---------------------------------------------------------------------------
// Stage 1: split-K GEMM via v_wmma_f32_16x16x32_bf16.
// Grid (4, 5, KSPLIT), 256 threads (8 waves). Block tile 128x128.
// Wave tile 32(M) x 64(N): 2 A-frags x 4 B-frags = 8 WMMAs / k-step.
// Ping-pong software pipeline with peeled tail.
// ---------------------------------------------------------------------------
__global__ __launch_bounds__(256)
void attr_gemm_wmma(const __bf16* __restrict__ feat,
                    const __bf16* __restrict__ W,
                    float* __restrict__ partial) {
  const int lane = threadIdx.x & 31;
  const int wave = threadIdx.x >> 5;
  const int lr   = lane & 15;
  const int half = lane >> 4;

  const int mBase = blockIdx.x * BM + (wave & 3) * 32;  // 4 waves along M
  const int nBase = blockIdx.y * BN + (wave >> 2) * 64; // 2 waves along N
  const int ks    = blockIdx.z;
  const long k0   = (long)ks * KCHUNK;

  // Wave-uniform scalar bases; fixed per-lane 32-bit offsets (fit easily:
  // max row offset ~15M elements).
  const __bf16* aBase = feat + k0;
  const __bf16* bBase = W + k0;
  const int aOff0 = (mBase + lr) * DSZ + half * 8;
  const int aOff1 = aOff0 + 16 * DSZ;
  int bOff[4];
#pragma unroll
  for (int j = 0; j < 4; ++j) {
    int c = nBase + j * 16 + lr;
    c = c < CSZ ? c : (CSZ - 1);  // clamp OOB classes (cols >=600 never read back)
    bOff[j] = c * DSZ + half * 8;
  }

  v8f acc[2][4];
#pragma unroll
  for (int i = 0; i < 2; ++i)
#pragma unroll
    for (int j = 0; j < 4; ++j) acc[i][j] = (v8f){0.f,0.f,0.f,0.f,0.f,0.f,0.f,0.f};

  // ping-pong fragment sets
  v16bf A0s[2], A1s[2], Bs[2][4];

#define LOAD_SET(s, kk)                                          \
  do {                                                           \
    A0s[s]   = load_frag_a(aBase, aOff0 + (kk));                 \
    A1s[s]   = load_frag_a(aBase, aOff1 + (kk));                 \
    Bs[s][0] = load_frag_b(bBase, bOff[0] + (kk));               \
    Bs[s][1] = load_frag_b(bBase, bOff[1] + (kk));               \
    Bs[s][2] = load_frag_b(bBase, bOff[2] + (kk));               \
    Bs[s][3] = load_frag_b(bBase, bOff[3] + (kk));               \
  } while (0)

#define MMA_SET(s)                                               \
  do {                                                           \
    _Pragma("unroll")                                            \
    for (int j = 0; j < 4; ++j) {                                \
      acc[0][j] = __builtin_amdgcn_wmma_f32_16x16x32_bf16(       \
          false, A0s[s], false, Bs[s][j], (short)0, acc[0][j],   \
          false, false);                                         \
      acc[1][j] = __builtin_amdgcn_wmma_f32_16x16x32_bf16(       \
          false, A1s[s], false, Bs[s][j], (short)0, acc[1][j],   \
          false, false);                                         \
    }                                                            \
  } while (0)

  LOAD_SET(0, 0);
  // main pipelined loop: 48 iterations cover k-steps 0..3040; all loads in range
  for (int kk = 0; kk < KCHUNK - 64; kk += 64) {
    __builtin_prefetch(aBase + aOff0 + kk + 128, 0, 1);  // global_prefetch_b8
    __builtin_prefetch(bBase + bOff[0] + kk + 128, 0, 1);
    LOAD_SET(1, kk + 32);
    MMA_SET(0);
    LOAD_SET(0, kk + 64);
    MMA_SET(1);
  }
  // tail: k-steps KCHUNK-64 (already in set0) and KCHUNK-32
  LOAD_SET(1, KCHUNK - 32);
  MMA_SET(0);
  MMA_SET(1);

#undef LOAD_SET
#undef MMA_SET

  // C/D layout: lanes 0-15 -> N=lane, M=vgpr; lanes 16-31 -> N=lane-16, M=8+vgpr.
#pragma unroll
  for (int i = 0; i < 2; ++i) {
#pragma unroll
    for (int j = 0; j < 4; ++j) {
      const int n  = nBase + j * 16 + lr;
      const int m0 = mBase + i * 16 + half * 8;
      float* outp = partial + ((long)ks * BSZ + m0) * NPAD + n;
#pragma unroll
      for (int r = 0; r < 8; ++r) outp[(long)r * NPAD] = acc[i][j][r];
    }
  }
}

// ---------------------------------------------------------------------------
// Stage 2: reduce K-splits, add bias, masked log-sigmoid loss, block-sum.
// Grid exactly covers B*C = 307200 elements with 1200 blocks of 256.
// ---------------------------------------------------------------------------
__global__ __launch_bounds__(256)
void attr_loss_partial(const float* __restrict__ partial,
                       const float* __restrict__ bias,
                       const int*   __restrict__ attr,
                       const float* __restrict__ mask,
                       float* __restrict__ blocksums) {
  const int idx = blockIdx.x * 256 + threadIdx.x;
  const int m = idx / CSZ;
  const int c = idx - m * CSZ;

  float s = bias[c];
#pragma unroll
  for (int ks = 0; ks < KSPLIT; ++ks)
    s += partial[((long)ks * BSZ + m) * NPAD + c];

  const float y = (float)attr[idx];
  const float w = mask[idx];
  // stable log-sigmoid: logsig(x) = min(x,0) - log1p(exp(-|x|))
  const float t      = __logf(1.f + __expf(-fabsf(s)));
  const float ls_pos = fminf(s, 0.f) - t;
  const float ls_neg = fminf(-s, 0.f) - t;
  float e = w * (y * ls_pos + (1.f - y) * ls_neg);

  __shared__ float red[256];
  red[threadIdx.x] = e;
  __syncthreads();
#pragma unroll
  for (int off = 128; off > 0; off >>= 1) {
    if (threadIdx.x < off) red[threadIdx.x] += red[threadIdx.x + off];
    __syncthreads();
  }
  if (threadIdx.x == 0) blocksums[blockIdx.x] = red[0];
}

// ---------------------------------------------------------------------------
// Stage 3: deterministic final reduction -> -mean
// ---------------------------------------------------------------------------
__global__ __launch_bounds__(256)
void attr_loss_final(const float* __restrict__ blocksums, int n,
                     float* __restrict__ out) {
  float s = 0.f;
  for (int i = threadIdx.x; i < n; i += 256) s += blocksums[i];
  __shared__ float red[256];
  red[threadIdx.x] = s;
  __syncthreads();
#pragma unroll
  for (int off = 128; off > 0; off >>= 1) {
    if (threadIdx.x < off) red[threadIdx.x] += red[threadIdx.x + off];
    __syncthreads();
  }
  if (threadIdx.x == 0) out[0] = -red[0] / (float)((long)BSZ * CSZ);
}

// ---------------------------------------------------------------------------
extern "C" void kernel_launch(void* const* d_in, const int* in_sizes, int n_in,
                              void* d_out, int out_size, void* d_ws, size_t ws_size,
                              hipStream_t stream) {
  (void)in_sizes; (void)n_in; (void)out_size; (void)ws_size;
  const float* feat = (const float*)d_in[0];  // [512, 25088] f32
  const float* W    = (const float*)d_in[1];  // [600, 25088] f32
  const float* bias = (const float*)d_in[2];  // [600] f32
  const int*   attr = (const int*)d_in[3];    // [512, 600] i32
  const float* mask = (const float*)d_in[4];  // [512, 600] f32
  float* out = (float*)d_out;                 // scalar loss

  // Workspace layout (all 16B aligned):
  //   featBf [12,845,056 bf16]  ~25.7 MB
  //   wBf    [15,052,800 bf16]  ~30.1 MB
  //   partial[8*512*640 f32]    ~10.5 MB
  //   blocksums[1200 f32]
  __bf16* featBf    = (__bf16*)d_ws;
  __bf16* wBf       = featBf + FEAT_N;
  float*  partial   = (float*)(wBf + W_N);
  float*  blocksums = partial + (long)KSPLIT * BSZ * NPAD;

  // Stage 0: convert (exact grids: FEAT_N/2048 = 6272, W_N/2048 = 7350)
  cvt_f32_bf16<<<(int)(FEAT_N / 2048), 256, 0, stream>>>(feat, featBf);
  cvt_f32_bf16<<<(int)(W_N    / 2048), 256, 0, stream>>>(W, wBf);

  // Stage 1: WMMA GEMM, split-K
  dim3 g1(BSZ / BM, NPAD / BN, KSPLIT);  // (4, 5, 8) = 160 workgroups
  attr_gemm_wmma<<<g1, 256, 0, stream>>>(featBf, wBf, partial);

  // Stage 2+3: fused loss + deterministic reduction
  const int blocks2 = (BSZ * CSZ) / 256;  // 1200 exact
  attr_loss_partial<<<blocks2, 256, 0, stream>>>(partial, bias, attr, mask, blocksums);
  attr_loss_final<<<1, 256, 0, stream>>>(blocksums, blocks2, out);
}
